// CNN_55422257988257
// MI455X (gfx1250) — matrix operand order repaired and checked by reading, not compiled, not run on hardware
//
#include <hip/hip_runtime.h>
#include <hip/hip_bf16.h>
#include <math.h>

// ---------------------------------------------------------------------------
// Types for CDNA5 WMMA
// ---------------------------------------------------------------------------
typedef __attribute__((ext_vector_type(16))) __bf16 v16bf;
typedef __attribute__((ext_vector_type(8)))  float  v8f;

union BfFrag {
    v16bf v;
    unsigned short u[16];
};

// round-to-nearest-even f32 -> bf16 (bit pattern)
__device__ __forceinline__ unsigned short f2bf(float f) {
    unsigned u = __float_as_uint(f);
    u += 0x7FFFu + ((u >> 16) & 1u);
    return (unsigned short)(u >> 16);
}

__device__ __forceinline__ float sigm(float x) {
    return 1.0f / (1.0f + __expf(-x));
}

#define HID 257

// ---------------------------------------------------------------------------
// Zero-init kernel (h/c state + zero vector + barrier counter).
// ---------------------------------------------------------------------------
__global__ void zero_kernel(float* __restrict__ p, int n) {
    int i = blockIdx.x * blockDim.x + threadIdx.x;
    if (i < n) p[i] = 0.0f;
}

// ---------------------------------------------------------------------------
// Generic VALID max-pool, NCHW, batch 1.
// ---------------------------------------------------------------------------
__global__ void pool_kernel(const float* __restrict__ X, float* __restrict__ Y,
                            int C, int IH, int IW, int OH, int OW, int k, int s) {
    int idx = blockIdx.x * blockDim.x + threadIdx.x;
    int tot = C * OH * OW;
    if (idx >= tot) return;
    int ow = idx % OW;
    int oh = (idx / OW) % OH;
    int c  = idx / (OW * OH);
    const float* base = X + (size_t)c * IH * IW + (size_t)(oh * s) * IW + ow * s;
    float m = -INFINITY;
    for (int r = 0; r < k; ++r)
        for (int q = 0; q < k; ++q) {
            float v = base[r * IW + q];
            m = v > m ? v : m;
        }
    Y[idx] = m;
}

// ---------------------------------------------------------------------------
// Implicit-GEMM conv (VALID, stride 1, batch 1) via v_wmma_f32_16x16x32_bf16.
//   M = Cout, N = OH*OW, K = Cin*KSZ*KSZ (KSZ compile-time).
//   One wave computes a 16(M) x 32(N) strip: A fragment reused for two B
//   fragments / two accumulators per K-chunk.
// ---------------------------------------------------------------------------
template <int KSZ>
__global__ __launch_bounds__(32) void conv_wmma_kernel(
    const float* __restrict__ X, const float* __restrict__ W,
    const float* __restrict__ Bias, float* __restrict__ Y,
    int Cin, int Cout, int IH, int IW, int OH, int OW)
{
    constexpr int KK = KSZ * KSZ;
    const int lane = threadIdx.x & 31;
    const int half = lane >> 4;
    const int mn   = lane & 15;
    const int Npos = OH * OW;
    const int Kdim = Cin * KK;
    const int m0   = blockIdx.y * 16;
    const int n0   = blockIdx.x * 32;

    int          ncol[2];
    bool         nvalid[2];
    const float* xpix[2];
    #pragma unroll
    for (int nt = 0; nt < 2; ++nt) {
        ncol[nt]   = n0 + nt * 16 + mn;
        nvalid[nt] = ncol[nt] < Npos;
        int oh = nvalid[nt] ? (ncol[nt] / OW) : 0;
        int ow = nvalid[nt] ? (ncol[nt] % OW) : 0;
        xpix[nt] = X + (size_t)oh * IW + ow;
    }

    v8f acc0 = {};
    v8f acc1 = {};

    for (int k0 = 0; k0 < Kdim; k0 += 32) {
        BfFrag a;
        const float* wrow = W + (size_t)(m0 + mn) * Kdim + k0 + half * 8;
        #pragma unroll
        for (int j = 0; j < 8; ++j) a.u[j]     = f2bf(wrow[j]);
        #pragma unroll
        for (int j = 0; j < 8; ++j) a.u[8 + j] = f2bf(wrow[16 + j]);

        BfFrag b[2];
        #pragma unroll
        for (int j = 0; j < 16; ++j) {
            const int kg  = k0 + half * 8 + (j & 7) + ((j >> 3) << 4);
            const int ci  = kg / KK;
            const int rem = kg - ci * KK;
            const int r   = rem / KSZ;
            const int cc  = rem - r * KSZ;
            const size_t off = (size_t)ci * IH * IW + r * IW + cc;
            #pragma unroll
            for (int nt = 0; nt < 2; ++nt) {
                float v = nvalid[nt] ? xpix[nt][off] : 0.0f;
                b[nt].u[j] = f2bf(v);
            }
        }

        acc0 = __builtin_amdgcn_wmma_f32_16x16x32_bf16(
            false, a.v, false, b[0].v, (short)0, acc0, false, false);
        acc1 = __builtin_amdgcn_wmma_f32_16x16x32_bf16(
            false, a.v, false, b[1].v, (short)0, acc1, false, false);
    }

    #pragma unroll
    for (int r = 0; r < 8; ++r) {
        const int   m   = m0 + r + 8 * half;
        const float bsv = Bias[m];
        if (nvalid[0]) {
            float v = acc0[r] + bsv;
            Y[(size_t)m * Npos + ncol[0]] = v > 0.0f ? v : 0.0f;
        }
        if (nvalid[1]) {
            float v = acc1[r] + bsv;
            Y[(size_t)m * Npos + ncol[1]] = v > 0.0f ? v : 0.0f;
        }
    }
}

// ---------------------------------------------------------------------------
// GEMV y = relu?(W x + b): one wave per row, float4 streaming (HBM-bound).
// ---------------------------------------------------------------------------
__global__ __launch_bounds__(256) void fc_kernel(
    const float* __restrict__ W, const float* __restrict__ bias,
    const float* __restrict__ x, float* __restrict__ y,
    int rows, int cols, int relu)
{
    int wave = (blockIdx.x * blockDim.x + threadIdx.x) >> 5;
    int lane = threadIdx.x & 31;
    if (wave >= rows) return;
    const float4* W4 = (const float4*)(W + (size_t)wave * cols);
    const float4* X4 = (const float4*)x;
    int c4n = cols >> 2;
    float s = 0.0f;
    for (int c = lane; c < c4n; c += 32) {
        __builtin_prefetch(&W4[c + 256], 0, 1);   // global_prefetch_b8
        float4 w  = W4[c];
        float4 xv = X4[c];
        s += w.x * xv.x + w.y * xv.y + w.z * xv.z + w.w * xv.w;
    }
    #pragma unroll
    for (int off = 16; off > 0; off >>= 1) s += __shfl_down(s, off, 32);
    if (lane == 0) {
        s += bias[wave];
        y[wave] = (relu && s < 0.0f) ? 0.0f : s;
    }
}

// ---------------------------------------------------------------------------
// Single LSTM cell kernel (used for the 4 encoder cells only).
// ---------------------------------------------------------------------------
__global__ __launch_bounds__(256) void lstm_cell_kernel(
    const float* __restrict__ Wih, const float* __restrict__ Whh,
    const float* __restrict__ bih, const float* __restrict__ bhh,
    const float* __restrict__ x, int xdim,
    const float* __restrict__ hin, float* __restrict__ hout,
    float* __restrict__ c, float* __restrict__ outrow)
{
    __shared__ float xh[HID + HID];
    __shared__ float gates[17 * 4];
    const int j0   = blockIdx.x * 17;
    const int nj   = min(17, HID - j0);
    const int tid  = threadIdx.x;
    const int lane = tid & 31;
    const int wave = tid >> 5;

    for (int t = tid; t < xdim; t += 256) xh[t] = x[t];
    for (int t = tid; t < HID;  t += 256) xh[xdim + t] = hin[t];
    __syncthreads();

    const int rows = nj * 4;
    for (int r = wave; r < rows; r += 8) {
        const int jj  = r >> 2;
        const int g   = r & 3;
        const int row = g * HID + (j0 + jj);
        const float* wi = Wih + (size_t)row * xdim;
        const float* wh = Whh + (size_t)row * HID;
        float acc = 0.0f;
        for (int t = lane; t < xdim; t += 32) acc += wi[t] * xh[t];
        for (int t = lane; t < HID;  t += 32) acc += wh[t] * xh[xdim + t];
        #pragma unroll
        for (int off = 16; off > 0; off >>= 1) acc += __shfl_down(acc, off, 32);
        if (lane == 0) gates[jj * 4 + g] = acc + bih[row] + bhh[row];
    }
    __syncthreads();

    for (int jj = tid; jj < nj; jj += 256) {
        const int j = j0 + jj;
        float gi = gates[jj * 4 + 0];
        float gf = gates[jj * 4 + 1];
        float gg = gates[jj * 4 + 2];
        float go = gates[jj * 4 + 3];
        float c2 = sigm(gf) * c[j] + sigm(gi) * tanhf(gg);
        float h2 = sigm(go) * tanhf(c2);
        c[j]    = c2;
        hout[j] = h2;
        if (outrow) outrow[j] = h2;
    }
}

// ---------------------------------------------------------------------------
// Persistent decoder: ONE kernel runs all 256 steps x 4 layers.
// 16 co-resident workgroups; generation-counter grid barrier between cells
// (atomicAdd release + acquire spin with s_sleep). Cell state c lives in
// registers for the whole decode. x||h staging uses the CDNA5 async path:
// global_load_async_to_lds_b32 + s_wait_asynccnt (ASYNCcnt-tracked).
// ---------------------------------------------------------------------------
struct DecParams {
    const float* Wih[4];
    const float* Whh[4];
    const float* bih[4];
    const float* bhh[4];
};

#define NWG 16

__device__ __forceinline__ void grid_barrier(int* cnt, int expected) {
    __syncthreads();
    if (threadIdx.x == 0) {
        __threadfence();                          // release this WG's stores
        atomicAdd(cnt, 1);
        while (__hip_atomic_load(cnt, __ATOMIC_ACQUIRE,
                                 __HIP_MEMORY_SCOPE_AGENT) < expected) {
            __builtin_amdgcn_s_sleep(1);
        }
        __threadfence();                          // acquire others' stores
    }
    __syncthreads();
}

__global__ __launch_bounds__(256) void decoder_kernel(
    DecParams P, const float* __restrict__ zvec,
    float* __restrict__ hbuf /* [4][2][257] */,
    const float* __restrict__ cinit /* [4][257] */,
    float* __restrict__ out /* [256][257] */,
    int* __restrict__ bcnt)
{
    __shared__ float xh[HID + HID];
    __shared__ float gates[17 * 4];
    const int j0   = blockIdx.x * 17;
    const int nj   = min(17, HID - j0);
    const int tid  = threadIdx.x;
    const int lane = tid & 31;
    const int wave = tid >> 5;

    // carried cell state (thread tid < nj owns hidden index j0+tid, 4 layers)
    float creg[4] = {0.f, 0.f, 0.f, 0.f};
    if (tid < nj) {
        #pragma unroll
        for (int li = 0; li < 4; ++li) creg[li] = cinit[li * HID + j0 + tid];
    }

    int phase = 0;
    for (int t = 0; t < 256; ++t) {
        const int pr = 1 - (t & 1);
        const int pw = t & 1;
        for (int li = 0; li < 4; ++li) {
            const float* x   = (li == 0)
                ? ((t == 0) ? zvec : out + (size_t)(t - 1) * HID)
                : hbuf + (size_t)((li - 1) * 2 + pw) * HID;
            const float* hin = hbuf + (size_t)(li * 2 + pr) * HID;
            float*       hout = hbuf + (size_t)(li * 2 + pw) * HID;

            // ---- stage x||hin into LDS via async-to-LDS (gfx1250 path) ----
            for (int s = tid; s < HID; s += 256) {
                unsigned           l0 = (unsigned)(size_t)&xh[s];
                unsigned long long g0 = (unsigned long long)(size_t)&x[s];
                asm volatile("global_load_async_to_lds_b32 %0, %1, off"
                             :: "v"(l0), "v"(g0) : "memory");
                unsigned           l1 = (unsigned)(size_t)&xh[HID + s];
                unsigned long long g1 = (unsigned long long)(size_t)&hin[s];
                asm volatile("global_load_async_to_lds_b32 %0, %1, off"
                             :: "v"(l1), "v"(g1) : "memory");
            }
            asm volatile("s_wait_asynccnt 0x0" ::: "memory");
            __syncthreads();

            // ---- gate rows: one row per wave, dot split across lanes ----
            const float* Wih = P.Wih[li];
            const float* Whh = P.Whh[li];
            const float* bih = P.bih[li];
            const float* bhh = P.bhh[li];
            const int rows = nj * 4;
            for (int r = wave; r < rows; r += 8) {
                const int jj  = r >> 2;
                const int g   = r & 3;
                const int row = g * HID + (j0 + jj);
                const float* wi = Wih + (size_t)row * HID;
                const float* wh = Whh + (size_t)row * HID;
                float acc = 0.0f;
                for (int q = lane; q < HID; q += 32) acc += wi[q] * xh[q];
                for (int q = lane; q < HID; q += 32) acc += wh[q] * xh[HID + q];
                #pragma unroll
                for (int off = 16; off > 0; off >>= 1) acc += __shfl_down(acc, off, 32);
                if (lane == 0) gates[jj * 4 + g] = acc + bih[row] + bhh[row];
            }
            __syncthreads();

            // ---- combine (WG-local slice; c carried in registers) ----
            if (tid < nj) {
                const int j = j0 + tid;
                float gi = gates[tid * 4 + 0];
                float gf = gates[tid * 4 + 1];
                float gg = gates[tid * 4 + 2];
                float go = gates[tid * 4 + 3];
                float c2 = sigm(gf) * creg[li] + sigm(gi) * tanhf(gg);
                float h2 = sigm(go) * tanhf(c2);
                creg[li] = c2;
                hout[j]  = h2;
                if (li == 3) out[(size_t)t * HID + j] = h2;
            }

            ++phase;
            grid_barrier(bcnt, NWG * phase);
        }
    }
}

// ---------------------------------------------------------------------------
// Host side
// ---------------------------------------------------------------------------
extern "C" void kernel_launch(void* const* d_in, const int* in_sizes, int n_in,
                              void* d_out, int out_size, void* d_ws, size_t ws_size,
                              hipStream_t stream)
{
    (void)in_sizes; (void)n_in; (void)out_size; (void)ws_size;

    const float* src = (const float*)d_in[0];
    float* out = (float*)d_out;

    float* ws   = (float*)d_ws;
    float* bufA = ws;               // 300000 floats
    float* bufB = ws + 300000;      // 300000 floats
    float* zvec = ws + 600000;      // 257 zeros
    float* hbuf = ws + 600257;      // h[4][2][257] = 2056
    float* cbuf = ws + 602313;      // c[4][257]   = 1028
    int*   bcnt = (int*)(ws + 603341);  // grid-barrier counter (1 int)

    // re-zero LSTM state + zero vector + barrier counter every call
    zero_kernel<<<(3342 + 255) / 256, 256, 0, stream>>>(zvec, 3342);

    // ---- CNN ----
    {
        int n = 32 * 32 * 32;
        pool_kernel<<<(n + 255) / 256, 256, 0, stream>>>(src, bufA, 32, 64, 64, 32, 32, 2, 2);
    }
    const int Ci[6] = {32, 64, 128, 256, 512, 512};
    const int Co[6] = {64, 128, 256, 512, 512, 512};
    const int Ks[6] = {3, 3, 3, 3, 3, 2};
    int ih = 32;
    float* cur = bufA;
    float* nxt = bufB;
    for (int li = 0; li < 6; ++li) {
        int oh   = ih - Ks[li] + 1;
        int Npos = oh * oh;
        dim3 grid((Npos + 31) / 32, Co[li] / 16);
        if (Ks[li] == 3) {
            conv_wmma_kernel<3><<<grid, 32, 0, stream>>>(
                cur, (const float*)d_in[1 + li], (const float*)d_in[7 + li], nxt,
                Ci[li], Co[li], ih, ih, oh, oh);
        } else {
            conv_wmma_kernel<2><<<grid, 32, 0, stream>>>(
                cur, (const float*)d_in[1 + li], (const float*)d_in[7 + li], nxt,
                Ci[li], Co[li], ih, ih, oh, oh);
        }
        ih = oh;
        float* t = cur; cur = nxt; nxt = t;
    }
    {
        int n = 512 * 10 * 10;
        pool_kernel<<<(n + 255) / 256, 256, 0, stream>>>(cur, nxt, 512, 21, 21, 10, 10, 3, 2);
        float* t = cur; cur = nxt; nxt = t;   // cur = [51200] features
    }

    // ---- FC head ----
    float* f1 = nxt;
    float* f2 = nxt + 2048;
    float* f3 = nxt + 2048 + 512;
    fc_kernel<<<2048 / 8, 256, 0, stream>>>((const float*)d_in[13], (const float*)d_in[16], cur, f1, 2048, 51200, 1);
    fc_kernel<<< 512 / 8, 256, 0, stream>>>((const float*)d_in[14], (const float*)d_in[17], f1,  f2,  512,  2048, 1);
    fc_kernel<<< 128 / 8, 256, 0, stream>>>((const float*)d_in[15], (const float*)d_in[18], f2,  f3,  128,   512, 0);

    // ---- encoder: 4 cells from zero state (read parity 0, write parity 1) ----
    auto Hbuf = [&](int li, int p) { return hbuf + (size_t)(li * 2 + p) * HID; };
    for (int li = 0; li < 4; ++li) {
        int b = 19 + li * 4;
        const float* x = (li == 0) ? f3 : Hbuf(li - 1, 1);
        int xdim = (li == 0) ? 128 : HID;
        lstm_cell_kernel<<<16, 256, 0, stream>>>(
            (const float*)d_in[b], (const float*)d_in[b + 1],
            (const float*)d_in[b + 2], (const float*)d_in[b + 3],
            x, xdim, Hbuf(li, 0), Hbuf(li, 1), cbuf + li * HID, nullptr);
    }

    // ---- decoder: one persistent kernel, 256 steps x 4 layers ----
    DecParams dp;
    for (int li = 0; li < 4; ++li) {
        int b = 35 + li * 4;
        dp.Wih[li] = (const float*)d_in[b];
        dp.Whh[li] = (const float*)d_in[b + 1];
        dp.bih[li] = (const float*)d_in[b + 2];
        dp.bhh[li] = (const float*)d_in[b + 3];
    }
    decoder_kernel<<<NWG, 256, 0, stream>>>(dp, zvec, hbuf, cbuf, out, bcnt);
}
